// FeatureExtractor_88476326298089
// MI455X (gfx1250) — matrix operand order repaired and checked by reading, not compiled, not run on hardware
//
#include <hip/hip_runtime.h>
#include <math.h>

typedef __attribute__((ext_vector_type(16))) __bf16 v16bf;
typedef __attribute__((ext_vector_type(8)))  float  v8f;

union FragBF {
  unsigned int u[8];
  uint4 q[2];
  v16bf v;
};

#define OUT_STRIDE 1891

__device__ __forceinline__ float geluf(float x) {
  return 0.5f * x * (1.0f + erff(x * 0.70710678118654752f));
}
__device__ __forceinline__ float symlogf(float x) {
  return copysignf(log1pf(fabsf(x)), x);
}
__device__ __forceinline__ unsigned int pack_bf16(float a, float b) {
  unsigned int ua = __float_as_uint(a);
  unsigned int ub = __float_as_uint(b);
  ua = (ua + 0x7FFFu + ((ua >> 16) & 1u)) >> 16;
  ub = (ub + 0x7FFFu + ((ub >> 16) & 1u)) >> 16;
  return (ua & 0xFFFFu) | (ub << 16);
}

// ---------------------------------------------------------------------------
// Precompute q = query @ qw for mob (256) and loot (64); batch-invariant.
// ---------------------------------------------------------------------------
__global__ void qprep_kernel(const float* __restrict__ mq, const float* __restrict__ mqw,
                             const float* __restrict__ lq, const float* __restrict__ lqw,
                             float* __restrict__ qmob, float* __restrict__ qloot) {
  int tid = threadIdx.x;
  if (tid < 256) {
    float s = 0.f;
    for (int k = 0; k < 256; ++k) s += mq[k] * mqw[k * 256 + tid];
    qmob[tid] = s;
  } else if (tid < 320) {
    int j = tid - 256;
    float s = 0.f;
    for (int k = 0; k < 64; ++k) s += lq[k] * lqw[k * 64 + j];
    qloot[j] = s;
  }
}

// ---------------------------------------------------------------------------
// Repack a row-major (K,N) f32 weight into bf16 wave32 B-fragment blocks.
// Block (kk,nt) = 32 lanes x 8 dwords; lane<16 holds K=kk*32+{0..7,16..23} (N=lane),
// lane>=16 holds K=kk*32+{8..15,24..31} (N=lane-16); dword j = packed K pair (table order).
// dst dword index == flat thread index (layouts coincide).
// ---------------------------------------------------------------------------
__global__ void repack_kernel(const float* __restrict__ src, unsigned int* __restrict__ dst,
                              int ksteps, int NT, int N) {
  int idx = blockIdx.x * 256 + threadIdx.x;
  int total = ksteps * NT * 256;
  if (idx >= total) return;
  int kk = idx / (NT * 256);
  int rem = idx - kk * (NT * 256);
  int nt = rem >> 8;
  int lj = rem & 255;
  int lane = lj >> 3, j = lj & 7;
  int n = nt * 16 + (lane & 15);
  int k0 = kk * 32 + ((lane & 16) ? 8 : 0) + ((j & 4) ? 16 : 0) + 2 * (j & 3);
  dst[idx] = pack_bf16(src[(size_t)k0 * N + n], src[(size_t)(k0 + 1) * N + n]);
}

// ---------------------------------------------------------------------------
// Mob pipeline: 16 samples / workgroup, 512 threads (16 waves, wave = sample).
// LDS: m tile (256 rows x 256 bf16) + pooled (16x256 f32) + q (256 f32).
// Per (sample, head): K/V tiles via v_wmma_f32_16x16x32_bf16, softmax on the
// C fragments with wave32 shuffles, then pooled @ pw with WMMA again.
// ---------------------------------------------------------------------------
__global__ __launch_bounds__(512) void mob_attn_kernel(
    const float* __restrict__ mobs, const float* __restrict__ mob_w,
    const float* __restrict__ mob_b, const unsigned int* __restrict__ kvwB,
    const unsigned int* __restrict__ pwB, const float* __restrict__ qvec,
    const float* __restrict__ pb, float* __restrict__ out) {
  extern __shared__ unsigned int smem[];
  unsigned int* ldsm = smem;                     // 256*128 dwords (bf16 m tile)
  float* ldspool = (float*)(smem + 256 * 128);   // 16*256
  float* ldsq = ldspool + 16 * 256;              // 256
  const int tid = threadIdx.x;
  const int wgBase = blockIdx.x * 16;

  if (tid < 256) ldsq[tid] = qvec[tid];

  {  // token MLP: m = gelu(f(mobs) @ mob_w + b), bf16 into LDS
    const int r = tid >> 1;                      // 0..255 = sample*16 + token
    const int s = r >> 4, t = r & 15;
    const int gb = wgBase + s;
    const float* mi = mobs + (size_t)(gb * 16 + t) * 4;
    const float f0 = symlogf(mi[0]);
    const float f1 = symlogf(mi[1]);
    const float f2 = mi[2] * 0.01f;
    const float f3 = mi[3];
    const int c0 = (tid & 1) * 128;
    for (int c = c0; c < c0 + 128; c += 2) {
      float v0 = geluf(f0 * mob_w[c]     + f1 * mob_w[256 + c]     + f2 * mob_w[512 + c]     + f3 * mob_w[768 + c]     + mob_b[c]);
      float v1 = geluf(f0 * mob_w[c + 1] + f1 * mob_w[256 + c + 1] + f2 * mob_w[512 + c + 1] + f3 * mob_w[768 + c + 1] + mob_b[c + 1]);
      ldsm[r * 128 + (c >> 1)] = pack_bf16(v0, v1);
    }
  }
  __syncthreads();

  const int wave = tid >> 5;
  const int lane = tid & 31;
  const int n15 = lane & 15;
  const int hi = lane >> 4;
  const int arow = wave * 16 + n15;  // A fragment row (token) for this sample

  for (int h = 0; h < 16; ++h) {
    v8f ck = {};
    v8f cv = {};
#pragma unroll
    for (int kk = 0; kk < 8; ++kk) {
      FragBF a, bk, bv;
      const int kb = kk * 32 + (hi ? 8 : 0);
      a.q[0] = *(const uint4*)(ldsm + arow * 128 + (kb >> 1));
      a.q[1] = *(const uint4*)(ldsm + arow * 128 + (kb >> 1) + 8);
      const uint4* pk = (const uint4*)(kvwB + (size_t)((kk * 32 + h) * 32 + lane) * 8);
      bk.q[0] = pk[0]; bk.q[1] = pk[1];
      const uint4* pv = (const uint4*)(kvwB + (size_t)((kk * 32 + 16 + h) * 32 + lane) * 8);
      bv.q[0] = pv[0]; bv.q[1] = pv[1];
      ck = __builtin_amdgcn_wmma_f32_16x16x32_bf16(false, a.v, false, bk.v, (short)0, ck, false, false);
      cv = __builtin_amdgcn_wmma_f32_16x16x32_bf16(false, a.v, false, bv.v, (short)0, cv, false, false);
    }
    // scores: reduce q*k over 16 feature lanes (tokens split across VGPR/halves)
    const float qv = ldsq[h * 16 + n15];
    float sc[8];
#pragma unroll
    for (int i = 0; i < 8; ++i) {
      float x = qv * ck[i];
      x += __shfl_xor(x, 1);
      x += __shfl_xor(x, 2);
      x += __shfl_xor(x, 4);
      x += __shfl_xor(x, 8);
      sc[i] = x * 0.25f;  // d=16 -> d^-0.5
    }
    float mx = sc[0];
#pragma unroll
    for (int i = 1; i < 8; ++i) mx = fmaxf(mx, sc[i]);
    mx = fmaxf(mx, __shfl_xor(mx, 16));
    float sum = 0.f;
#pragma unroll
    for (int i = 0; i < 8; ++i) { sc[i] = expf(sc[i] - mx); sum += sc[i]; }
    sum += __shfl_xor(sum, 16);
    const float rs = 1.0f / sum;
    float acc = 0.f;
#pragma unroll
    for (int i = 0; i < 8; ++i) acc += sc[i] * rs * cv[i];
    acc += __shfl_xor(acc, 16);
    if (lane < 16) ldspool[wave * 256 + h * 16 + lane] = acc;
  }
  __syncthreads();

  {  // output projection: pooled (16x256) @ pw (256x256), wave = n-tile
    const int nt = wave;
    v8f c = {};
#pragma unroll
    for (int kk = 0; kk < 8; ++kk) {
      FragBF a, b;
      const int kb = kk * 32 + (hi ? 8 : 0);
      const float* prow = ldspool + n15 * 256;
#pragma unroll
      for (int jj = 0; jj < 4; ++jj) {
        a.u[jj]     = pack_bf16(prow[kb + 2 * jj],      prow[kb + 2 * jj + 1]);
        a.u[4 + jj] = pack_bf16(prow[kb + 16 + 2 * jj], prow[kb + 16 + 2 * jj + 1]);
      }
      const uint4* pB = (const uint4*)(pwB + (size_t)((kk * 16 + nt) * 32 + lane) * 8);
      b.q[0] = pB[0]; b.q[1] = pB[1];
      c = __builtin_amdgcn_wmma_f32_16x16x32_bf16(false, a.v, false, b.v, (short)0, c, false, false);
    }
    const float bias = pb[nt * 16 + n15];
#pragma unroll
    for (int i = 0; i < 8; ++i) {
      const int row = i + (hi ? 8 : 0);
      out[(size_t)(wgBase + row) * OUT_STRIDE + 1408 + nt * 16 + n15] = geluf(c[i] + bias);
    }
  }
}

// ---------------------------------------------------------------------------
// Loot pipeline: same structure, D=64, 8 heads of d=8 (head pair per n-tile).
// ---------------------------------------------------------------------------
__global__ __launch_bounds__(512) void loot_attn_kernel(
    const float* __restrict__ loot, const float* __restrict__ loot_w,
    const float* __restrict__ loot_b, const unsigned int* __restrict__ kvwB,
    const unsigned int* __restrict__ pwB, const float* __restrict__ qvec,
    const float* __restrict__ pb, float* __restrict__ out) {
  __shared__ unsigned int ldsl[256 * 32];  // 256 rows x 64 bf16
  __shared__ float ldspool[16 * 64];
  __shared__ float ldsq[64];
  const int tid = threadIdx.x;
  const int wgBase = blockIdx.x * 16;
  if (tid < 64) ldsq[tid] = qvec[tid];

  {
    const int r = tid >> 1;
    const int s = r >> 4, t = r & 15;
    const int gb = wgBase + s;
    const float* li = loot + (size_t)(gb * 16 + t) * 3;
    const float f0 = symlogf(li[0]);
    const float f1 = symlogf(li[1]);
    const float f2 = li[2];
    const int c0 = (tid & 1) * 32;
    for (int c = c0; c < c0 + 32; c += 2) {
      float v0 = geluf(f0 * loot_w[c]     + f1 * loot_w[64 + c]     + f2 * loot_w[128 + c]     + loot_b[c]);
      float v1 = geluf(f0 * loot_w[c + 1] + f1 * loot_w[64 + c + 1] + f2 * loot_w[128 + c + 1] + loot_b[c + 1]);
      ldsl[r * 32 + (c >> 1)] = pack_bf16(v0, v1);
    }
  }
  __syncthreads();

  const int wave = tid >> 5;
  const int lane = tid & 31;
  const int n15 = lane & 15;
  const int hi = lane >> 4;
  const int arow = wave * 16 + n15;

  for (int nt = 0; nt < 4; ++nt) {  // n-tile = head pair (2nt, 2nt+1)
    v8f ck = {};
    v8f cv = {};
#pragma unroll
    for (int kk = 0; kk < 2; ++kk) {
      FragBF a, bk, bv;
      const int kb = kk * 32 + (hi ? 8 : 0);
      a.q[0] = *(const uint4*)(ldsl + arow * 32 + (kb >> 1));
      a.q[1] = *(const uint4*)(ldsl + arow * 32 + (kb >> 1) + 8);
      const uint4* pk = (const uint4*)(kvwB + (size_t)((kk * 8 + nt) * 32 + lane) * 8);
      bk.q[0] = pk[0]; bk.q[1] = pk[1];
      const uint4* pv = (const uint4*)(kvwB + (size_t)((kk * 8 + 4 + nt) * 32 + lane) * 8);
      bv.q[0] = pv[0]; bv.q[1] = pv[1];
      ck = __builtin_amdgcn_wmma_f32_16x16x32_bf16(false, a.v, false, bk.v, (short)0, ck, false, false);
      cv = __builtin_amdgcn_wmma_f32_16x16x32_bf16(false, a.v, false, bv.v, (short)0, cv, false, false);
    }
    const float qv = ldsq[nt * 16 + n15];
    float sc[8];
#pragma unroll
    for (int i = 0; i < 8; ++i) {  // reduce over d=8 lanes per head group
      float x = qv * ck[i];
      x += __shfl_xor(x, 1);
      x += __shfl_xor(x, 2);
      x += __shfl_xor(x, 4);
      sc[i] = x * 0.35355339059327373f;  // d=8 -> d^-0.5
    }
    float mx = sc[0];
#pragma unroll
    for (int i = 1; i < 8; ++i) mx = fmaxf(mx, sc[i]);
    mx = fmaxf(mx, __shfl_xor(mx, 16));  // xor16 keeps head, flips token half
    float sum = 0.f;
#pragma unroll
    for (int i = 0; i < 8; ++i) { sc[i] = expf(sc[i] - mx); sum += sc[i]; }
    sum += __shfl_xor(sum, 16);
    const float rs = 1.0f / sum;
    float acc = 0.f;
#pragma unroll
    for (int i = 0; i < 8; ++i) acc += sc[i] * rs * cv[i];
    acc += __shfl_xor(acc, 16);
    if (lane < 16) ldspool[wave * 64 + nt * 16 + lane] = acc;
  }
  __syncthreads();

  if (wave < 4) {  // pooled (16x64) @ pw (64x64)
    const int nt = wave;
    v8f c = {};
#pragma unroll
    for (int kk = 0; kk < 2; ++kk) {
      FragBF a, b;
      const int kb = kk * 32 + (hi ? 8 : 0);
      const float* prow = ldspool + n15 * 64;
#pragma unroll
      for (int jj = 0; jj < 4; ++jj) {
        a.u[jj]     = pack_bf16(prow[kb + 2 * jj],      prow[kb + 2 * jj + 1]);
        a.u[4 + jj] = pack_bf16(prow[kb + 16 + 2 * jj], prow[kb + 16 + 2 * jj + 1]);
      }
      const uint4* pB = (const uint4*)(pwB + (size_t)((kk * 4 + nt) * 32 + lane) * 8);
      b.q[0] = pB[0]; b.q[1] = pB[1];
      c = __builtin_amdgcn_wmma_f32_16x16x32_bf16(false, a.v, false, b.v, (short)0, c, false, false);
    }
    const float bias = pb[nt * 16 + n15];
#pragma unroll
    for (int i = 0; i < 8; ++i) {
      const int row = i + (hi ? 8 : 0);
      out[(size_t)(wgBase + row) * OUT_STRIDE + 1664 + nt * 16 + n15] = geluf(c[i] + bias);
    }
  }
}

// ---------------------------------------------------------------------------
// All remaining features: one 256-thread block per sample (streaming / VALU).
// ---------------------------------------------------------------------------
__global__ __launch_bounds__(256) void misc_kernel(
    const int* __restrict__ topmat, const float* __restrict__ theights,
    const float* __restrict__ invst, const float* __restrict__ ppos,
    const int* __restrict__ prot, const float* __restrict__ hp,
    const float* __restrict__ timep, const float* __restrict__ amask,
    const float* __restrict__ status, const float* __restrict__ disc,
    const int* __restrict__ pact, const float* __restrict__ block_w,
    const float* __restrict__ block_b, const float* __restrict__ chw,
    const float* __restrict__ chb, const float* __restrict__ cvw,
    const float* __restrict__ cvb, const float* __restrict__ inv_w,
    const float* __restrict__ inv_b, const float* __restrict__ pa_w,
    const float* __restrict__ pa_b, float* __restrict__ out) {
  __shared__ float mat[225 * 8];
  __shared__ float th[81];
  __shared__ float linv[27];
  const int b = blockIdx.x;
  const int tid = threadIdx.x;
  float* o = out + (size_t)b * OUT_STRIDE;

  for (int idx = tid; idx < 1800; idx += 256) {  // one-hot @ block_w = row gather
    const int cell = idx >> 3, ch = idx & 7;
    const int mid = topmat[b * 225 + cell];
    mat[idx] = geluf(block_w[mid * 8 + ch] + block_b[ch]);
  }
  for (int idx = tid; idx < 81; idx += 256) {
    const int y = idx / 9, x = idx - y * 9;
    th[idx] = theights[b * 225 + (3 + y) * 15 + (3 + x)] * 0.25f;
  }
  if (tid < 27) linv[tid] = log1pf(invst[b * 27 + tid]);
  __syncthreads();

  if (tid < 200) {  // near: mat[5:10,5:10]
    const int rr = tid / 40, rem = tid - rr * 40;
    const int cc = rem >> 3, ch = rem & 7;
    o[tid] = mat[((5 + rr) * 15 + 5 + cc) * 8 + ch];
  }
  if (tid < 200) {  // 3x3 max pool, channel-major
    const int ch = tid / 25, rem = tid - ch * 25;
    const int i = rem / 5, j = rem - i * 5;
    float mv = -1e30f;
    for (int a = 0; a < 3; ++a)
      for (int c2 = 0; c2 < 3; ++c2)
        mv = fmaxf(mv, mat[((i * 3 + a) * 15 + j * 3 + c2) * 8 + ch]);
    o[200 + tid] = mv;
  }
  for (int idx = tid; idx < 504; idx += 256) {  // horizontal 1x3 conv (8,9,7)
    const int ch = idx / 63, rem = idx - ch * 63;
    const int y = rem / 7, x = rem - y * 7;
    const float s = th[y * 9 + x] * chw[ch * 3] + th[y * 9 + x + 1] * chw[ch * 3 + 1] +
                    th[y * 9 + x + 2] * chw[ch * 3 + 2] + chb[ch];
    o[400 + idx] = geluf(s);
  }
  for (int idx = tid; idx < 504; idx += 256) {  // vertical 3x1 conv (8,7,9)
    const int ch = idx / 63, rem = idx - ch * 63;
    const int y = rem / 9, x = rem - y * 9;
    const float s = th[y * 9 + x] * cvw[ch * 3] + th[(y + 1) * 9 + x] * cvw[ch * 3 + 1] +
                    th[(y + 2) * 9 + x] * cvw[ch * 3 + 2] + cvb[ch];
    o[904 + idx] = geluf(s);
  }
  if (tid < 54) {  // inventory MLP
    float s = inv_b[tid];
    for (int i = 0; i < 27; ++i) s += linv[i] * inv_w[i * 54 + tid];
    o[1728 + tid] = geluf(s);
  }
  if (tid == 0) {  // flat scalars
    o[1782] = symlogf(ppos[b * 3]);
    o[1783] = symlogf(ppos[b * 3 + 1]);
    o[1784] = ppos[b * 3 + 2] * 0.25f;
    const int rot = prot[b];
    for (int k = 0; k < 4; ++k) o[1785 + k] = (k == rot) ? 1.f : 0.f;
    o[1789] = hp[b] * 0.01f;
    const float t = log1pf(timep[b]) * (1.0f / 7.0f);
    o[1790] = t;
    o[1791] = fmodf(t, 1.f);
    o[1792] = fmodf(t, 2.f) * 0.5f;
  }
  if (tid < 39) o[1793 + tid] = amask[b * 39 + tid];
  if (tid < 4) o[1832 + tid] = log1pf(status[b * 4 + tid]);
  if (tid < 39) o[1836 + tid] = disc[b * 39 + tid];
  if (tid < 16) {  // past-action embedding sum
    float s = pa_b[tid];
    for (int p = 0; p < 5; ++p) {
      const int a = pact[b * 5 + p];
      s += pa_w[(p * 39 + a) * 16 + tid];
    }
    o[1875 + tid] = geluf(s);
  }
}

extern "C" void kernel_launch(void* const* d_in, const int* in_sizes, int n_in,
                              void* d_out, int out_size, void* d_ws, size_t ws_size,
                              hipStream_t stream) {
  (void)in_sizes; (void)n_in; (void)out_size; (void)ws_size;
  const int*   topmat    = (const int*)d_in[0];
  const float* theights  = (const float*)d_in[1];
  const float* mobs      = (const float*)d_in[2];
  const float* loot      = (const float*)d_in[3];
  const float* invst     = (const float*)d_in[4];
  const float* ppos      = (const float*)d_in[5];
  const int*   prot      = (const int*)d_in[6];
  const float* hp        = (const float*)d_in[7];
  const float* timep     = (const float*)d_in[8];
  const float* amask     = (const float*)d_in[9];
  const float* status    = (const float*)d_in[10];
  const float* disc      = (const float*)d_in[11];
  const int*   pact      = (const int*)d_in[12];
  const float* block_w   = (const float*)d_in[13];
  const float* block_b   = (const float*)d_in[14];
  const float* chw       = (const float*)d_in[15];
  const float* chb       = (const float*)d_in[16];
  const float* cvw       = (const float*)d_in[17];
  const float* cvb       = (const float*)d_in[18];
  const float* mob_w     = (const float*)d_in[19];
  const float* mob_b     = (const float*)d_in[20];
  const float* mob_query = (const float*)d_in[21];
  const float* mob_qw    = (const float*)d_in[22];
  const float* mob_kvw   = (const float*)d_in[23];
  const float* mob_pw    = (const float*)d_in[24];
  const float* mob_pb    = (const float*)d_in[25];
  const float* loot_w    = (const float*)d_in[26];
  const float* loot_b    = (const float*)d_in[27];
  const float* loot_query= (const float*)d_in[28];
  const float* loot_qw   = (const float*)d_in[29];
  const float* loot_kvw  = (const float*)d_in[30];
  const float* loot_pw   = (const float*)d_in[31];
  const float* loot_pb   = (const float*)d_in[32];
  const float* inv_w     = (const float*)d_in[33];
  const float* inv_b     = (const float*)d_in[34];
  const float* pa_w      = (const float*)d_in[35];
  const float* pa_b      = (const float*)d_in[36];
  float* out = (float*)d_out;

  char* ws = (char*)d_ws;
  float* qmob  = (float*)(ws);                               // 256 f32
  float* qloot = (float*)(ws + 1024);                        // 64 f32
  unsigned int* kvwm = (unsigned int*)(ws + 4096);           // 8*32 blocks  (256 KB)
  unsigned int* pwm  = (unsigned int*)(ws + 4096 + 262144);  // 8*16 blocks  (128 KB)
  unsigned int* kvwl = (unsigned int*)(ws + 4096 + 262144 + 131072);          // 2*8  (16 KB)
  unsigned int* pwl  = (unsigned int*)(ws + 4096 + 262144 + 131072 + 16384);  // 2*4  (8 KB)

  qprep_kernel<<<1, 320, 0, stream>>>(mob_query, mob_qw, loot_query, loot_qw, qmob, qloot);
  repack_kernel<<<256, 256, 0, stream>>>(mob_kvw, kvwm, 8, 32, 512);
  repack_kernel<<<128, 256, 0, stream>>>(mob_pw,  pwm,  8, 16, 256);
  repack_kernel<<<16,  256, 0, stream>>>(loot_kvw, kvwl, 2, 8, 128);
  repack_kernel<<<8,   256, 0, stream>>>(loot_pw,  pwl,  2, 4, 64);

  const int nBlocks = 8192 / 16;
  const size_t mobSmem = (size_t)(256 * 128 + 16 * 256 + 256) * 4;  // 148480 B
  mob_attn_kernel<<<nBlocks, 512, mobSmem, stream>>>(mobs, mob_w, mob_b, kvwm, pwm, qmob, mob_pb, out);
  loot_attn_kernel<<<nBlocks, 512, 0, stream>>>(loot, loot_w, loot_b, kvwl, pwl, qloot, loot_pb, out);
  misc_kernel<<<8192, 256, 0, stream>>>(topmat, theights, invst, ppos, prot, hp, timep,
                                        amask, status, disc, pact, block_w, block_b,
                                        chw, chb, cvw, cvb, inv_w, inv_b, pa_w, pa_b, out);
}
